// LoFTRAttention_2516850835970
// MI455X (gfx1250) — compile-verified, matrix operand order checked by reading, not run on hardware
//
#include <hip/hip_runtime.h>
#include <hip/hip_bf16.h>
#include <stdint.h>

#define D_MODEL 1024
#define NHEAD   16
#define D_HEAD  64
#define BSZ     2
#define LQ      2048
#define LK      2048
#define QK_SCALE 0.125f   // 64^-0.5

typedef __attribute__((ext_vector_type(16))) __bf16         v16bf;
typedef __attribute__((ext_vector_type(8)))  float          v8f;
typedef __attribute__((ext_vector_type(16))) unsigned short u16x16;

// D = A(16x32 bf16) * B(32x16 bf16) + C(16x16 f32)
__device__ __forceinline__ v8f wmma_bf16(v16bf a, v16bf b, v8f c) {
  return __builtin_amdgcn_wmma_f32_16x16x32_bf16(false, a, false, b, (short)0, c,
                                                 false, false);
}

__device__ __forceinline__ v16bf ld_bf16x16(const unsigned short* p) {
  return __builtin_bit_cast(v16bf, *(const u16x16*)p);
}

__device__ __forceinline__ unsigned short f2bf(float f) {
  unsigned u = __float_as_uint(f);
  u += 0x7FFFu + ((u >> 16) & 1u);   // round-to-nearest-even
  return (unsigned short)(u >> 16);
}

// CDNA5 async DMA: 16B global -> LDS, tracked by ASYNCcnt (no VGPR data path).
__device__ __forceinline__ void async_g2l_b128(unsigned lds_off, const void* gaddr) {
  asm volatile("global_load_async_to_lds_b128 %0, %1, off"
               :: "v"(lds_off), "v"(gaddr) : "memory");
}
__device__ __forceinline__ void wait_asynccnt0() {
  asm volatile("s_wait_asynccnt 0x0" ::: "memory");
}
__device__ __forceinline__ void wait_dscnt0() {
  asm volatile("s_wait_dscnt 0x0" ::: "memory");
}

// ---------------------------------------------------------------- fp32 -> bf16
__global__ __launch_bounds__(256) void cvt_bf16(const float* __restrict__ x,
                                                unsigned short* __restrict__ y,
                                                int n) {
  for (int i = blockIdx.x * blockDim.x + threadIdx.x; i < n;
       i += gridDim.x * blockDim.x)
    y[i] = f2bf(x[i]);
}

// ---------------------------------------------------------------- GEMM
// C[M,N] = A[M,Kd] @ W[N,Kd]^T   (bf16 in, f32 WMMA accumulate)
// mode 0: store bf16 row-major [M,N]
// mode 1: store bf16 as Vt[b, h, d, l2]  (per-head transposed V)
// mode 2: store f32 row-major [M,N]
// Block: 256 threads (8 waves), tile 128 rows x 128 cols, K-step 32.
// W tile double-buffered in LDS, filled by async-to-LDS DMA.
__global__ __launch_bounds__(256) void gemm_bf16(const unsigned short* __restrict__ A,
                                                 const unsigned short* __restrict__ W,
                                                 void* __restrict__ Cout,
                                                 int M, int N, int Kd, int mode) {
  __shared__ unsigned short ldsB[2][128 * 32];  // 2 x 8 KB: [n_local][k_local]

  const int lane  = threadIdx.x & 31;
  const int wvid  = threadIdx.x >> 5;   // wave 0..7 -> 16-row band
  const int lcol  = lane & 15;
  const int lhalf = lane >> 4;          // k-half (A/B) or row-half (C)
  const int row0  = blockIdx.x * 128 + wvid * 16;
  const int col0  = blockIdx.y * 128;

  // Async fill: 8 KB tile = 512 x 16B chunks; each of 256 threads moves 2 chunks.
  // chunk c -> n_local = c>>2, k-chunk kc = c&3 (8 halves each).
  auto fill = [&](int buf, int k0) {
#pragma unroll
    for (int cc = 0; cc < 2; cc++) {
      int c  = threadIdx.x + cc * 256;
      int nl = c >> 2, kc = c & 3;
      const unsigned short* g = W + (size_t)(col0 + nl) * Kd + k0 + kc * 8;
      unsigned lo = (unsigned)(unsigned long long)&ldsB[buf][nl * 32 + kc * 8];
      async_g2l_b128(lo, g);
    }
  };

  v8f acc[8] = {};

  fill(0, 0);
  wait_asynccnt0();
  __syncthreads();

  int buf = 0;
  for (int k0 = 0; k0 < Kd; k0 += 32) {
    if (k0 + 32 < Kd) {
      fill(buf ^ 1, k0 + 32);  // DMA next tile while computing on this one
      __builtin_prefetch(A + (size_t)(row0 + lcol) * Kd + k0 + 32 + lhalf * 16, 0, 1);
    }

    v16bf af = ld_bf16x16(A + (size_t)(row0 + lcol) * Kd + k0 + lhalf * 16);
    v16bf bfrag[8];
#pragma unroll
    for (int n0 = 0; n0 < 8; n0++)
      bfrag[n0] = ld_bf16x16(&ldsB[buf][(n0 * 16 + lcol) * 32 + lhalf * 16]);
#pragma unroll
    for (int n0 = 0; n0 < 8; n0++)        // back-to-back WMMA burst
      acc[n0] = wmma_bf16(af, bfrag[n0], acc[n0]);

    wait_asynccnt0();   // next tile resident
    __syncthreads();    // everyone done reading current tile
    buf ^= 1;
  }

  // C layout: VGPR r, lanes 0-15 -> row r, lanes 16-31 -> row r+8; col = lane%16
  if (mode == 0) {
    unsigned short* C = (unsigned short*)Cout;
#pragma unroll
    for (int n0 = 0; n0 < 8; n0++)
#pragma unroll
      for (int r = 0; r < 8; r++) {
        int row = row0 + r + 8 * lhalf;
        int col = col0 + n0 * 16 + lcol;
        C[(size_t)row * N + col] = f2bf(acc[n0][r]);
      }
  } else if (mode == 1) {
    unsigned short* C = (unsigned short*)Cout;  // Vt[b][h][d][l2]
#pragma unroll
    for (int n0 = 0; n0 < 8; n0++)
#pragma unroll
      for (int r = 0; r < 8; r++) {
        int row = row0 + r + 8 * lhalf;   // b*LK + l2
        int col = col0 + n0 * 16 + lcol;  // h*64 + d
        int bi = row >> 11, l2 = row & 2047;
        int hh = col >> 6,  dd = col & 63;
        C[((((size_t)bi * NHEAD + hh) * D_HEAD + dd) << 11) + l2] = f2bf(acc[n0][r]);
      }
  } else {
    float* C = (float*)Cout;
#pragma unroll
    for (int n0 = 0; n0 < 8; n0++)
#pragma unroll
      for (int r = 0; r < 8; r++) {
        int row = row0 + r + 8 * lhalf;
        int col = col0 + n0 * 16 + lcol;
        C[(size_t)row * N + col] = acc[n0][r];
      }
  }
}

// ---------------------------------------------------------------- flash attention
// One wave per 16-row Q tile of one (b,h); streams K/V in 32-key steps.
// Q,K: bf16 [B*L, 1024] (head-major cols). Vt: bf16 [B,H,64,L2]. Ctx: bf16 [B*L,1024].
// P-tile C->A relayout goes through a wave-private LDS patch; DS ops are in-order
// per wave, so a wave-level s_wait_dscnt suffices (no workgroup barriers).
__global__ __launch_bounds__(128) void attn_kernel(const unsigned short* __restrict__ Q,
                                                   const unsigned short* __restrict__ Km,
                                                   const unsigned short* __restrict__ Vt,
                                                   unsigned short* __restrict__ Ctx) {
  __shared__ unsigned short ldsP[4][16 * 32];  // per-wave P-tile staging

  const int lane  = threadIdx.x & 31;
  const int wv    = threadIdx.x >> 5;
  const int lcol  = lane & 15;
  const int lhalf = lane >> 4;

  const int tile = blockIdx.x * 4 + wv;   // 4096 tiles total
  const int bh   = tile >> 7;             // 0..31
  const int qt   = tile & 127;            // q-tile within (b,h)
  const int bi   = bh >> 4;
  const int h    = bh & 15;
  const int qrow0 = qt * 16;

  // Q A-fragments (d-halves 0/1), loaded once
  const unsigned short* Qp =
      Q + (size_t)(bi * LQ + qrow0 + lcol) * D_MODEL + h * D_HEAD + lhalf * 16;
  v16bf qf0 = ld_bf16x16(Qp);
  v16bf qf1 = ld_bf16x16(Qp + 32);

  float mrun[8], lrun[8];
#pragma unroll
  for (int r = 0; r < 8; r++) { mrun[r] = -3.0e38f; lrun[r] = 0.f; }
  v8f o0 = {}, o1 = {}, o2 = {}, o3 = {};

  unsigned short* pbuf = &ldsP[wv][0];
  const unsigned short* Kbh = Km + (size_t)(bi * LK) * D_MODEL + h * D_HEAD;
  const unsigned short* Vbh = Vt + ((size_t)(bi * NHEAD + h)) * D_HEAD * LK;

  for (int l2b = 0; l2b < LK; l2b += 32) {
    // ---- S = Q @ K^T for 32 keys (two 16x16 column blocks)
    const unsigned short* K0 = Kbh + (size_t)(l2b + lcol) * D_MODEL + lhalf * 16;
    const unsigned short* K1 = Kbh + (size_t)(l2b + 16 + lcol) * D_MODEL + lhalf * 16;
    v16bf b00 = ld_bf16x16(K0);
    v16bf b01 = ld_bf16x16(K0 + 32);
    v16bf b10 = ld_bf16x16(K1);
    v16bf b11 = ld_bf16x16(K1 + 32);
    v8f s0 = {}, s1 = {};
    s0 = wmma_bf16(qf0, b00, s0);
    s0 = wmma_bf16(qf1, b01, s0);
    s1 = wmma_bf16(qf0, b10, s1);
    s1 = wmma_bf16(qf1, b11, s1);

    // ---- streaming softmax; row r+8*lhalf spread over 16 lanes of a group
#pragma unroll
    for (int r = 0; r < 8; r++) {
      float a0 = s0[r] * QK_SCALE;
      float a1 = s1[r] * QK_SCALE;
      float mx = fmaxf(a0, a1);
#pragma unroll
      for (int off = 8; off >= 1; off >>= 1) mx = fmaxf(mx, __shfl_xor(mx, off, 32));
      float mnew = fmaxf(mrun[r], mx);
      float p0 = __expf(a0 - mnew);
      float p1 = __expf(a1 - mnew);
      float sum = p0 + p1;
#pragma unroll
      for (int off = 8; off >= 1; off >>= 1) sum += __shfl_xor(sum, off, 32);
      float alpha = __expf(mrun[r] - mnew);
      mrun[r] = mnew;
      lrun[r] = lrun[r] * alpha + sum;
      o0[r] *= alpha; o1[r] *= alpha; o2[r] *= alpha; o3[r] *= alpha;
      int mrow = r + 8 * lhalf;
      pbuf[mrow * 32 + lcol]      = f2bf(p0);
      pbuf[mrow * 32 + 16 + lcol] = f2bf(p1);
    }
    wait_dscnt0();  // wave-private staging: all P stores visible to this wave

    // ---- O += P @ V : P re-read as A-fragment; V B-frags contiguous from Vt
    v16bf pf = ld_bf16x16(&pbuf[lcol * 32 + lhalf * 16]);
    const unsigned short* Vp = Vbh + (size_t)lcol * LK + l2b + lhalf * 16;
    v16bf vb0 = ld_bf16x16(Vp);
    v16bf vb1 = ld_bf16x16(Vp + 16 * LK);
    v16bf vb2 = ld_bf16x16(Vp + 32 * LK);
    v16bf vb3 = ld_bf16x16(Vp + 48 * LK);
    o0 = wmma_bf16(pf, vb0, o0);
    o1 = wmma_bf16(pf, vb1, o1);
    o2 = wmma_bf16(pf, vb2, o2);
    o3 = wmma_bf16(pf, vb3, o3);
  }

  // ---- normalize and store context (bf16, head-major columns)
#pragma unroll
  for (int r = 0; r < 8; r++) {
    float inv = 1.0f / lrun[r];
    size_t row = (size_t)(bi * LQ + qrow0 + r + 8 * lhalf);
    unsigned short* Cp = Ctx + row * D_MODEL + h * D_HEAD + lcol;
    Cp[0]  = f2bf(o0[r] * inv);
    Cp[16] = f2bf(o1[r] * inv);
    Cp[32] = f2bf(o2[r] * inv);
    Cp[48] = f2bf(o3[r] * inv);
  }
}

// ----------------------------------------------------------------
extern "C" void kernel_launch(void* const* d_in, const int* in_sizes, int n_in,
                              void* d_out, int out_size, void* d_ws, size_t ws_size,
                              hipStream_t stream) {
  const float* x   = (const float*)d_in[0];
  const float* src = (const float*)d_in[1];
  const float* Wq  = (const float*)d_in[2];
  const float* Wk  = (const float*)d_in[3];
  const float* Wv  = (const float*)d_in[4];
  const float* Wo  = (const float*)d_in[5];

  char* ws = (char*)d_ws;
  const size_t MB = 1024ull * 1024ull;
  unsigned short* xb  = (unsigned short*)(ws + 0 * MB);   // 8 MB
  unsigned short* sb  = (unsigned short*)(ws + 8 * MB);   // 8 MB
  unsigned short* wqb = (unsigned short*)(ws + 16 * MB);  // 2 MB
  unsigned short* wkb = (unsigned short*)(ws + 18 * MB);  // 2 MB
  unsigned short* wvb = (unsigned short*)(ws + 20 * MB);  // 2 MB
  unsigned short* wob = (unsigned short*)(ws + 22 * MB);  // 2 MB
  unsigned short* Qb  = (unsigned short*)(ws + 24 * MB);  // 8 MB
  unsigned short* Kb  = (unsigned short*)(ws + 32 * MB);  // 8 MB
  unsigned short* Vtb = (unsigned short*)(ws + 40 * MB);  // 8 MB (per-head transposed)
  unsigned short* Ctx = (unsigned short*)(ws + 48 * MB);  // 8 MB

  const int nx = BSZ * LQ * D_MODEL;       // 4,194,304
  const int nw = D_MODEL * D_MODEL;        // 1,048,576

  cvt_bf16<<<dim3(4096), dim3(256), 0, stream>>>(x, xb, nx);
  cvt_bf16<<<dim3(4096), dim3(256), 0, stream>>>(src, sb, nx);
  cvt_bf16<<<dim3(1024), dim3(256), 0, stream>>>(Wq, wqb, nw);
  cvt_bf16<<<dim3(1024), dim3(256), 0, stream>>>(Wk, wkb, nw);
  cvt_bf16<<<dim3(1024), dim3(256), 0, stream>>>(Wv, wvb, nw);
  cvt_bf16<<<dim3(1024), dim3(256), 0, stream>>>(Wo, wob, nw);

  const int M = BSZ * LQ;  // 4096
  dim3 gg(M / 128, D_MODEL / 128);  // (32, 8)
  gemm_bf16<<<gg, 256, 0, stream>>>(xb, wqb, (void*)Qb,  M, D_MODEL, D_MODEL, 0);
  gemm_bf16<<<gg, 256, 0, stream>>>(sb, wkb, (void*)Kb,  M, D_MODEL, D_MODEL, 0);
  gemm_bf16<<<gg, 256, 0, stream>>>(sb, wvb, (void*)Vtb, M, D_MODEL, D_MODEL, 1);

  // 4096 q-tiles, 4 waves/block
  attn_kernel<<<dim3(1024), dim3(128), 0, stream>>>(Qb, Kb, Vtb, Ctx);

  gemm_bf16<<<gg, 256, 0, stream>>>(Ctx, wob, d_out, M, D_MODEL, D_MODEL, 2);
}
